// MoE_20899310862533
// MI455X (gfx1250) — matrix operand order repaired and checked by reading, not compiled, not run on hardware
//
#include <hip/hip_runtime.h>
#include <hip/hip_bf16.h>

typedef unsigned short u16;
typedef unsigned int   u32;
typedef __attribute__((ext_vector_type(16))) __bf16 v16bf;
typedef __attribute__((ext_vector_type(8)))  float  v8f;

union BFrag { uint4 u4[2]; v16bf v; };

__device__ __forceinline__ u16 f2bf(float f) {
    union { float f; u32 u; } c; c.f = f;
    u32 u = c.u + 0x7FFFu + ((c.u >> 16) & 1u);   // round-to-nearest-even
    return (u16)(u >> 16);
}

__device__ __forceinline__ v8f wmma_bf16(v16bf a, v16bf b, v8f c) {
    return __builtin_amdgcn_wmma_f32_16x16x32_bf16(
        /*neg_a=*/false, a, /*neg_b=*/false, b,
        /*c_mod=*/(short)0, c, /*reuse_a=*/false, /*reuse_b=*/false);
}

// B-fragment: 32x16 bf16 tile stored as one 1024B chunk in fragment-lane order.
// lane l<16: N=l, K=0..15 ; lane l>=16: N=l-16, K=16..31 (16 bf16 per lane).
__device__ __forceinline__ v16bf load_bfrag(const u16* __restrict__ chunk, int lane) {
    const uint4* p = (const uint4*)chunk;
    BFrag f;
    f.u4[0] = p[lane * 2];
    f.u4[1] = p[lane * 2 + 1];
    return f.v;
}

// A-fragment from LDS, row-major bf16, row stride STRIDE elements.
// STRIDE chosen so (STRIDE/2) % 64 == 4 dwords -> lanes hit banks 4m..4m+3,
// conflict-free across the 16-lane half.
template<int STRIDE>
__device__ __forceinline__ v16bf load_afrag(const u16* A, int m, int k0, int lane) {
    const uint4* p = (const uint4*)(A + m * STRIDE + k0 + ((lane >= 16) ? 16 : 0));
    BFrag f;
    f.u4[0] = p[0];
    f.u4[1] = p[1];
    return f.v;
}

// ---------------------------------------------------------------------------
// Pack: f32 [K,N] row-major -> bf16 fragment chunks (512 elems / 32x16 tile).
// chunk c = kt*(N/16) + nt ; element t = lane*16 + j
// ---------------------------------------------------------------------------
__global__ void pack_bf16_chunks(const float* __restrict__ src, u16* __restrict__ dst,
                                 int N, int total) {
    int idx = blockIdx.x * blockDim.x + threadIdx.x;
    if (idx >= total) return;
    int c = idx >> 9;
    int t = idx & 511;
    int ntiles = N >> 4;
    int nt = c % ntiles;
    int kt = c / ntiles;
    int lane = t >> 4;
    int j = t & 15;
    int k = kt * 32 + ((lane >= 16) ? 16 : 0) + j;
    int n = nt * 16 + (lane & 15);
    dst[idx] = f2bf(src[k * N + n]);
}

// ---------------------------------------------------------------------------
// Gating MLP: 256 -> 128 -> 256 -> 128 -> 16, leaky relu, softmax.
// One block = 32 batch rows, 8 waves, WMMA-tiled, activations in LDS (bf16).
// Activation buffers padded to stride 264 (bank-conflict-free A fragments).
// ---------------------------------------------------------------------------
#define GSTR 264

template<int K, int N>
__device__ __forceinline__ void gate_layer(const u16* A, u16* Out,
                                           const u16* __restrict__ pB,
                                           const float* __restrict__ bias,
                                           int lane, int w) {
    constexpr int NT = N / 16;
    constexpr int TILES = 2 * NT;
    constexpr int TPW = (TILES + 7) / 8;
    int hi = (lane >= 16) ? 8 : 0;
    for (int it = 0; it < TPW; ++it) {
        int t = w * TPW + it;
        int nt = t >> 1, mt = t & 1;
        if (nt >= NT) break;
        v8f acc = {0.f, 0.f, 0.f, 0.f, 0.f, 0.f, 0.f, 0.f};
        int m = mt * 16 + (lane & 15);
        for (int kt = 0; kt < K / 32; ++kt) {
            v16bf a = load_afrag<GSTR>(A, m, kt * 32, lane);
            v16bf b = load_bfrag(pB + (kt * NT + nt) * 512, lane);
            acc = wmma_bf16(a, b, acc);
        }
        int col = nt * 16 + (lane & 15);
        float bv = bias[col];
        int rowbase = mt * 16 + hi;
        #pragma unroll
        for (int i = 0; i < 8; ++i) {
            float v = acc[i] + bv;
            v = (v >= 0.f) ? v : 0.01f * v;       // LeakyReLU(0.01)
            Out[(rowbase + i) * GSTR + col] = f2bf(v);
        }
    }
}

__global__ void __launch_bounds__(256) moe_gating_kernel(
    const float* __restrict__ x,
    const u16* __restrict__ pWg1, const float* __restrict__ bg1,
    const u16* __restrict__ pWg2, const float* __restrict__ bg2,
    const u16* __restrict__ pWg3, const float* __restrict__ bg3,
    const u16* __restrict__ pWg4, const float* __restrict__ bg4,
    float* __restrict__ gates) {
    __shared__ __align__(16) u16 bufA[32 * GSTR];
    __shared__ __align__(16) u16 bufB[32 * GSTR];
    __shared__ float logitsS[32 * 16];

    const int tid = threadIdx.x;
    const int lane = tid & 31;
    const int w = tid >> 5;
    const int rb = blockIdx.x * 32;

    // stage x tile (f32 -> bf16), 64 float4 per row
    const float4* x4 = (const float4*)(x + rb * 256);
    for (int i = tid; i < 2048; i += 256) {
        float4 v = x4[i];
        int row = i >> 6, c4 = i & 63;
        ushort4 p;
        p.x = f2bf(v.x); p.y = f2bf(v.y); p.z = f2bf(v.z); p.w = f2bf(v.w);
        *(ushort4*)&bufA[row * GSTR + c4 * 4] = p;
    }
    __syncthreads();

    gate_layer<256, 128>(bufA, bufB, pWg1, bg1, lane, w);
    __syncthreads();
    gate_layer<128, 256>(bufB, bufA, pWg2, bg2, lane, w);
    __syncthreads();
    gate_layer<256, 128>(bufA, bufB, pWg3, bg3, lane, w);
    __syncthreads();

    // layer 4: K=128, N=16 -> f32 logits (waves 0 and 1 only)
    if (w < 2) {
        int mt = w & 1;
        int hi = (lane >= 16) ? 8 : 0;
        int m = mt * 16 + (lane & 15);
        v8f acc = {0.f, 0.f, 0.f, 0.f, 0.f, 0.f, 0.f, 0.f};
        for (int kt = 0; kt < 4; ++kt) {
            v16bf a = load_afrag<GSTR>(bufB, m, kt * 32, lane);
            v16bf b = load_bfrag(pWg4 + kt * 512, lane);
            acc = wmma_bf16(a, b, acc);
        }
        int col = lane & 15;
        float bv = bg4[col];
        #pragma unroll
        for (int i = 0; i < 8; ++i)
            logitsS[(mt * 16 + i + hi) * 16 + col] = acc[i] + bv;
    }
    __syncthreads();

    // softmax over E=16, one lane per row
    if (tid < 32) {
        float v[16], mx = -3.4e38f;
        #pragma unroll
        for (int e = 0; e < 16; ++e) { v[e] = logitsS[tid * 16 + e]; mx = fmaxf(mx, v[e]); }
        float s = 0.f;
        #pragma unroll
        for (int e = 0; e < 16; ++e) { v[e] = __expf(v[e] - mx); s += v[e]; }
        float inv = 1.f / s;
        #pragma unroll
        for (int e = 0; e < 16; ++e) gates[(rb + tid) * 16 + e] = v[e] * inv;
    }
}

// ---------------------------------------------------------------------------
// Fused experts: per 64-row tile (512 threads = 16 waves, 4(m) x 4(n)):
//   eh = relu(X@We1[e]+be1[e]) * gate[:,e]   (bf16 in LDS, padded stride)
//   out_acc += eh @ We2[e]                   (persistent f32 regs)
// final: out = out_acc + sum_e gate*be2[e]
// ---------------------------------------------------------------------------
#define XSTR 264   // 256 + 8 pad  (row = 528 B, dword stride % 64 == 4)
#define HSTR 520   // 512 + 8 pad  (row = 1040 B, dword stride % 64 == 4)

__global__ void __launch_bounds__(512) moe_experts_kernel(
    const float* __restrict__ x,
    const u16* __restrict__ pWe1, const float* __restrict__ be1,
    const u16* __restrict__ pWe2, const float* __restrict__ be2,
    const float* __restrict__ gates, float* __restrict__ out) {
    __shared__ __align__(16) u16 Xs[64 * XSTR];    // ~33 KB
    __shared__ __align__(16) u16 EHs[64 * HSTR];   // ~65 KB
    __shared__ float gs[64 * 16];                  // 4 KB
    __shared__ float be1s[512];                    // 2 KB

    const int tid = threadIdx.x;
    const int lane = tid & 31;
    const int w = tid >> 5;           // 0..15
    const int rb = blockIdx.x * 64;

    // stage x tile (bf16, padded rows) and gates
    const float4* x4 = (const float4*)(x + (size_t)rb * 256);
    for (int i = tid; i < 4096; i += 512) {        // 64 rows x 64 float4
        float4 v = x4[i];
        int row = i >> 6, c4 = i & 63;
        ushort4 p;
        p.x = f2bf(v.x); p.y = f2bf(v.y); p.z = f2bf(v.z); p.w = f2bf(v.w);
        *(ushort4*)&Xs[row * XSTR + c4 * 4] = p;
    }
    for (int i = tid; i < 1024; i += 512) gs[i] = gates[rb * 16 + i];

    const int mt  = w >> 2;            // 0..3 : 16-row m tile
    const int ns1 = (w & 3) * 8;       // GEMM1: 8 n-tiles (128 cols of H)
    const int ns2 = (w & 3) * 4;       // GEMM2: 4 n-tiles (64 cols of O)
    const int m   = mt * 16 + (lane & 15);
    const int hi  = (lane >= 16) ? 8 : 0;

    v8f oacc[4];
    #pragma unroll
    for (int j = 0; j < 4; ++j) oacc[j] = {0.f, 0.f, 0.f, 0.f, 0.f, 0.f, 0.f, 0.f};

    for (int e = 0; e < 16; ++e) {
        __syncthreads();   // Xs/gs staged (e=0); EHs free from previous GEMM2
        be1s[tid < 512 ? tid : 0] = be1[e * 512 + tid];

        // ---- GEMM1: [64,256] x [256,512] ----
        v8f acc1[8];
        #pragma unroll
        for (int j = 0; j < 8; ++j) acc1[j] = {0.f, 0.f, 0.f, 0.f, 0.f, 0.f, 0.f, 0.f};
        const u16* pW1e = pWe1 + (size_t)e * 8 * 32 * 512;
        for (int kt = 0; kt < 8; ++kt) {
            if (kt < 7)
                __builtin_prefetch(pW1e + ((kt + 1) * 32 + ns1) * 512, 0, 0);
            v16bf a = load_afrag<XSTR>(Xs, m, kt * 32, lane);
            #pragma unroll
            for (int j = 0; j < 8; ++j) {
                v16bf b = load_bfrag(pW1e + (kt * 32 + ns1 + j) * 512, lane);
                acc1[j] = wmma_bf16(a, b, acc1[j]);
            }
        }
        __syncthreads();   // be1s visible

        // gate fragment for this wave's rows
        float gf[8];
        #pragma unroll
        for (int i = 0; i < 8; ++i) gf[i] = gs[(mt * 16 + i + hi) * 16 + e];

        // epilogue: +be1, relu, *gate, bf16 -> LDS
        #pragma unroll
        for (int j = 0; j < 8; ++j) {
            int col = (ns1 + j) * 16 + (lane & 15);
            float bv = be1s[col];
            #pragma unroll
            for (int i = 0; i < 8; ++i) {
                float v = acc1[j][i] + bv;
                v = (v > 0.f) ? v : 0.f;
                EHs[(mt * 16 + i + hi) * HSTR + col] = f2bf(v * gf[i]);
            }
        }
        __syncthreads();   // EHs ready

        // ---- GEMM2: [64,512] x [512,256], accumulate across experts ----
        const u16* pW2e = pWe2 + (size_t)e * 16 * 16 * 512;
        for (int kt = 0; kt < 16; ++kt) {
            if (kt < 15)
                __builtin_prefetch(pW2e + ((kt + 1) * 16 + ns2) * 512, 0, 0);
            v16bf a = load_afrag<HSTR>(EHs, m, kt * 32, lane);
            #pragma unroll
            for (int j = 0; j < 4; ++j) {
                v16bf b = load_bfrag(pW2e + (kt * 16 + ns2 + j) * 512, lane);
                oacc[j] = wmma_bf16(a, b, oacc[j]);
            }
        }
    }

    // final epilogue: + sum_e gate*be2[e], store f32
    #pragma unroll
    for (int j = 0; j < 4; ++j) {
        int col = (ns2 + j) * 16 + (lane & 15);
        float bev[16];
        #pragma unroll
        for (int e2 = 0; e2 < 16; ++e2) bev[e2] = be2[e2 * 256 + col];
        #pragma unroll
        for (int i = 0; i < 8; ++i) {
            int row = mt * 16 + i + hi;
            float b = 0.f;
            #pragma unroll
            for (int e2 = 0; e2 < 16; ++e2) b += gs[row * 16 + e2] * bev[e2];
            out[(size_t)(rb + row) * 256 + col] = oacc[j][i] + b;
        }
    }
}

// ---------------------------------------------------------------------------
extern "C" void kernel_launch(void* const* d_in, const int* in_sizes, int n_in,
                              void* d_out, int out_size, void* d_ws, size_t ws_size,
                              hipStream_t stream) {
    const float* x   = (const float*)d_in[0];
    const float* We1 = (const float*)d_in[1];
    const float* be1 = (const float*)d_in[2];
    const float* We2 = (const float*)d_in[3];
    const float* be2 = (const float*)d_in[4];
    const float* Wg1 = (const float*)d_in[5];
    const float* bg1 = (const float*)d_in[6];
    const float* Wg2 = (const float*)d_in[7];
    const float* bg2 = (const float*)d_in[8];
    const float* Wg3 = (const float*)d_in[9];
    const float* bg3 = (const float*)d_in[10];
    const float* Wg4 = (const float*)d_in[11];
    const float* bg4 = (const float*)d_in[12];
    float* out = (float*)d_out;

    char* ws = (char*)d_ws;
    float* gates = (float*)ws;  ws += (size_t)32768 * 16 * 4;
    u16* pWe1 = (u16*)ws;       ws += (size_t)16 * 256 * 512 * 2;
    u16* pWe2 = (u16*)ws;       ws += (size_t)16 * 512 * 256 * 2;
    u16* pWg1 = (u16*)ws;       ws += (size_t)256 * 128 * 2;
    u16* pWg2 = (u16*)ws;       ws += (size_t)128 * 256 * 2;
    u16* pWg3 = (u16*)ws;       ws += (size_t)256 * 128 * 2;
    u16* pWg4 = (u16*)ws;       ws += (size_t)128 * 16 * 2;

    auto pack = [&](const float* src, u16* dst, int K, int N) {
        int total = K * N;
        pack_bf16_chunks<<<(total + 255) / 256, 256, 0, stream>>>(src, dst, N, total);
    };
    pack(We1, pWe1, 16 * 256, 512);   // expert k-tiles contiguous per expert
    pack(We2, pWe2, 16 * 512, 256);
    pack(Wg1, pWg1, 256, 128);
    pack(Wg2, pWg2, 128, 256);
    pack(Wg3, pWg3, 256, 128);
    pack(Wg4, pWg4, 128, 16);

    moe_gating_kernel<<<32768 / 32, 256, 0, stream>>>(
        x, pWg1, bg1, pWg2, bg2, pWg3, bg3, pWg4, bg4, gates);
    moe_experts_kernel<<<32768 / 64, 512, 0, stream>>>(
        x, pWe1, be1, pWe2, be2, gates, out);
}
